// MultiHeadAttention_44710609551642
// MI455X (gfx1250) — matrix operand order repaired and checked by reading, not compiled, not run on hardware
//
#include <hip/hip_runtime.h>

// ---------------------------------------------------------------------------
// MHA forward for MI455X (gfx1250), wave32 + v_wmma_f32_16x16x32_bf16.
// d_out = [out (4096*512 f32)] ++ [attn (2*8*2048*2048 f32)]
// d_ws  layout: Qh bf16 4MB | Kh bf16 4MB | VhT bf16 4MB | concat f32 8MB
// Attention kernel: 4 waves cooperate on one 16x2048 LDS score strip.
// ---------------------------------------------------------------------------

typedef __attribute__((ext_vector_type(16))) __bf16 v16bf;
typedef __attribute__((ext_vector_type(8)))  float  v8f;

#define D_MODEL 512
#define SEQ     2048
#define NHEAD   8
#define DEPTH   64
#define BATCH   2
#define MROWS   (BATCH * SEQ)     // 4096
#define LDSROW  (SEQ + 4)         // stride % 64 banks == 4 -> conflict-free

__device__ __forceinline__ v8f wmma_bf16(v16bf a, v16bf b, v8f c) {
  return __builtin_amdgcn_wmma_f32_16x16x32_bf16(false, a, false, b,
                                                 (short)0, c, false, false);
}

// ----- fragment loaders (contiguous k = 16*half + e in both A and B) -----------
// A (16x32): lane lm = row M; element e: K = 16*half + e
__device__ __forceinline__ v16bf load_a_f32(const float* src, int ld, int row,
                                            int kb, int lm, int half) {
  const float* p0 = src + (size_t)(row + lm) * ld + kb + 16 * half;
  const float* p1 = p0 + 8;
  v16bf a;
#pragma unroll
  for (int i = 0; i < 8; ++i) { a[i] = (__bf16)p0[i]; a[i + 8] = (__bf16)p1[i]; }
  return a;
}

__device__ __forceinline__ v16bf load_a_bf16(const __bf16* src, int ld, int row,
                                             int kb, int lm, int half) {
  const __bf16* p0 = src + (size_t)(row + lm) * ld + kb + 16 * half;
  const __bf16* p1 = p0 + 8;
  v16bf a;
#pragma unroll
  for (int i = 0; i < 8; ++i) { a[i] = p0[i]; a[i + 8] = p1[i]; }
  return a;
}

// B (32x16): lane lm = col N; element e: K = kb + 16*half + e.
// Row-major [K x N] fp32 source (weights): strided by ld.
__device__ __forceinline__ v16bf load_b_f32(const float* src, int ld, int kb,
                                            int col, int lm, int half) {
  const float* p = src + (size_t)(kb + 16 * half) * ld + col + lm;
  v16bf b;
#pragma unroll
  for (int i = 0; i < 16; ++i) b[i] = (__bf16)p[(size_t)i * ld];
  return b;
}

// B (32x16) from [N x K] row-major bf16 (K contiguous per lane -> 32B loads).
__device__ __forceinline__ v16bf load_b_kcontig_bf16(const __bf16* src, int ld,
                                                     int kb, int col, int lm,
                                                     int half) {
  const __bf16* p = src + (size_t)(col + lm) * ld + kb + 16 * half;
  v16bf b;
#pragma unroll
  for (int i = 0; i < 16; ++i) b[i] = p[i];
  return b;
}

// ---------------------------------------------------------------------------
// Kernel 1: fused QKV projections.  8 waves/block, one 16x64 tile per wave.
// Q (pre-scaled by 1/sqrt(depth)), K -> [BH, S, 64] bf16 ;
// V -> transposed [BH, 64, S] bf16.
// grid = 3 * 2048 tiles / 8 = 768 blocks.
// ---------------------------------------------------------------------------
__global__ __launch_bounds__(256) void qkv_proj_kernel(
    const float* __restrict__ q, const float* __restrict__ k,
    const float* __restrict__ v, const float* __restrict__ wq,
    const float* __restrict__ bq, const float* __restrict__ wk,
    const float* __restrict__ bk, const float* __restrict__ wv,
    const float* __restrict__ bv, __bf16* __restrict__ Qh,
    __bf16* __restrict__ Kh, __bf16* __restrict__ VhT) {
  const int lane = threadIdx.x & 31, lm = lane & 15, half = lane >> 4;
  const int wid = threadIdx.x >> 5;
  const int g = blockIdx.x * 8 + wid;      // global tile id, 0..6143
  const int which = g >> 11;               // 0=q 1=k 2=v (2048 tiles each)
  const int tile  = g & 2047;
  const int bm = tile >> 3, bn = tile & 7; // 256 x 8
  const int rowBase = bm * 16, colBase = bn * 64;  // head h == bn

  const float* X  = (which == 0) ? q  : (which == 1) ? k  : v;
  const float* W  = (which == 0) ? wq : (which == 1) ? wk : wv;
  const float* Bv = (which == 0) ? bq : (which == 1) ? bk : bv;

  v8f zero = {};
  v8f c[4] = {zero, zero, zero, zero};
  for (int kt = 0; kt < D_MODEL; kt += 32) {
    v16bf a = load_a_f32(X, D_MODEL, rowBase, kt, lm, half);
#pragma unroll
    for (int t = 0; t < 4; ++t) {
      v16bf b = load_b_f32(W, D_MODEL, kt, colBase + 16 * t, lm, half);
      c[t] = wmma_bf16(a, b, c[t]);
    }
  }

  const int h = bn;
  const float qscale = (which == 0) ? 0.125f : 1.0f;  // fold 1/sqrt(64) into Q
#pragma unroll
  for (int t = 0; t < 4; ++t) {
    const int d = 16 * t + lm;
    const float bias = Bv[colBase + d];
#pragma unroll
    for (int r = 0; r < 8; ++r) {
      const int m = rowBase + r + 8 * half;  // 0..4095
      const int bb = m >> 11, s = m & (SEQ - 1);
      const float val = (c[t][r] + bias) * qscale;
      const size_t bh = (size_t)bb * NHEAD + h;
      if (which == 0)
        Qh[(bh * SEQ + s) * DEPTH + d] = (__bf16)val;
      else if (which == 1)
        Kh[(bh * SEQ + s) * DEPTH + d] = (__bf16)val;
      else
        VhT[(bh * DEPTH + d) * SEQ + s] = (__bf16)val;
    }
  }
}

// ---------------------------------------------------------------------------
// Kernel 2: fused attention.  grid = BH(16) * S/16(128); 4 waves per block.
//   phase 1: logits -> LDS strip (n-tiles split across waves)
//   phase 2: softmax rows (4 rows per wave), attn streamed to d_out
//   phase 3: ctx = P x V, split-K across waves, LDS partial reduction
// ---------------------------------------------------------------------------
__global__ __launch_bounds__(128) void attn_kernel(
    const __bf16* __restrict__ Qh, const __bf16* __restrict__ Kh,
    const __bf16* __restrict__ VhT, float* __restrict__ attn,
    float* __restrict__ concat) {
  __shared__ float sP[16 * LDSROW];    // 131,328 B score strip
  __shared__ float sC[4 * 16 * DEPTH]; // 16,384 B ctx partials

  const int lane = threadIdx.x & 31, lm = lane & 15, half = lane >> 4;
  const int wid = threadIdx.x >> 5;    // 0..3
  const int bh = blockIdx.x >> 7, qt = blockIdx.x & 127;
  const __bf16* Q = Qh + (size_t)bh * SEQ * DEPTH;
  const __bf16* K = Kh + (size_t)bh * SEQ * DEPTH;
  const __bf16* V = VhT + (size_t)bh * DEPTH * SEQ;
  const int rowBase = qt * 16;

  // ---- phase 1: logits = Qs (16x64) . K^T (64 x S) ----
  const v16bf a0 = load_a_bf16(Q, DEPTH, rowBase, 0, lm, half);
  const v16bf a1 = load_a_bf16(Q, DEPTH, rowBase, 32, lm, half);
  for (int nt = wid * 32; nt < wid * 32 + 32; ++nt) {
    // prefetch next K tile (16 rows x 64 bf16 = 2KB) toward WGP
    __builtin_prefetch(K + (size_t)(nt + 1) * 16 * DEPTH + lane * 64, 0, 1);
    v16bf b0 = load_b_kcontig_bf16(K, DEPTH, 0, nt * 16, lm, half);
    v16bf b1 = load_b_kcontig_bf16(K, DEPTH, 32, nt * 16, lm, half);
    v8f c = {};
    c = wmma_bf16(a0, b0, c);
    c = wmma_bf16(a1, b1, c);
#pragma unroll
    for (int r = 0; r < 8; ++r)
      sP[(r + 8 * half) * LDSROW + nt * 16 + lm] = c[r];
  }
  __syncthreads();

  // ---- phase 2: row softmax (wave w owns rows 4w..4w+3) ----
  for (int j = 0; j < 4; ++j) {
    const int m = wid * 4 + j;
    float* row = sP + m * LDSROW;
    float mx = -3.402823466e38f;
    for (int i = lane; i < SEQ; i += 32) mx = fmaxf(mx, row[i]);
#pragma unroll
    for (int off = 16; off; off >>= 1) mx = fmaxf(mx, __shfl_xor(mx, off, 32));
    float sum = 0.f;
    for (int i = lane; i < SEQ; i += 32) {
      float e = __expf(row[i] - mx);
      row[i] = e;
      sum += e;
    }
#pragma unroll
    for (int off = 16; off; off >>= 1) sum += __shfl_xor(sum, off, 32);
    const float rs = 1.f / sum;
    float* arow = attn + ((size_t)bh * SEQ + rowBase + m) * SEQ;
    for (int i = lane; i < SEQ; i += 32) {
      float p = row[i] * rs;
      row[i] = p;
      arow[i] = p;  // coalesced 128B stores
    }
  }
  __syncthreads();

  // ---- phase 3: ctx (16x64) = P (16xS) . V (Sx64), split-K over waves ----
  v8f zero = {};
  v8f c2[4] = {zero, zero, zero, zero};
  const int k0 = wid * (SEQ / 4), k1 = k0 + (SEQ / 4);
  for (int kt = k0; kt < k1; kt += 32) {
    __builtin_prefetch(V + (size_t)lane * SEQ + kt + 32, 0, 1);
    v16bf a = load_a_f32(sP, LDSROW, 0, kt, lm, half);
#pragma unroll
    for (int t = 0; t < 4; ++t) {
      v16bf b = load_b_kcontig_bf16(V, SEQ, kt, 16 * t, lm, half);
      c2[t] = wmma_bf16(a, b, c2[t]);
    }
  }
  float* myC = sC + wid * 16 * DEPTH;
#pragma unroll
  for (int t = 0; t < 4; ++t)
#pragma unroll
    for (int r = 0; r < 8; ++r)
      myC[(r + 8 * half) * DEPTH + 16 * t + lm] = c2[t][r];
  __syncthreads();

  // reduce 4 partials (1024 elements / 128 threads) and write concat
  const int bb = bh >> 3, h = bh & 7;
  for (int idx = threadIdx.x; idx < 16 * DEPTH; idx += 128) {
    float s = sC[idx] + sC[16 * DEPTH + idx] + sC[2 * 16 * DEPTH + idx] +
              sC[3 * 16 * DEPTH + idx];
    const int rrow = idx >> 6, col = idx & (DEPTH - 1);
    concat[((size_t)bb * SEQ + rowBase + rrow) * D_MODEL + h * DEPTH + col] = s;
  }
}

// ---------------------------------------------------------------------------
// Kernel 3: out = concat @ wo_w + wo_b.  8 waves/block; grid = 2048/8 = 256.
// ---------------------------------------------------------------------------
__global__ __launch_bounds__(256) void out_proj_kernel(
    const float* __restrict__ X, const float* __restrict__ W,
    const float* __restrict__ bias, float* __restrict__ out) {
  const int lane = threadIdx.x & 31, lm = lane & 15, half = lane >> 4;
  const int wid = threadIdx.x >> 5;
  const int g = blockIdx.x * 8 + wid;   // 0..2047
  const int bm = g >> 3, bn = g & 7;
  const int rowBase = bm * 16, colBase = bn * 64;

  v8f zero = {};
  v8f c[4] = {zero, zero, zero, zero};
  for (int kt = 0; kt < D_MODEL; kt += 32) {
    v16bf a = load_a_f32(X, D_MODEL, rowBase, kt, lm, half);
#pragma unroll
    for (int t = 0; t < 4; ++t) {
      v16bf b = load_b_f32(W, D_MODEL, kt, colBase + 16 * t, lm, half);
      c[t] = wmma_bf16(a, b, c[t]);
    }
  }
#pragma unroll
  for (int t = 0; t < 4; ++t) {
    const float bv = bias[colBase + 16 * t + lm];
#pragma unroll
    for (int r = 0; r < 8; ++r) {
      const int m = rowBase + r + 8 * half;
      out[(size_t)m * D_MODEL + colBase + 16 * t + lm] = c[t][r] + bv;
    }
  }
}

// ---------------------------------------------------------------------------
extern "C" void kernel_launch(void* const* d_in, const int* in_sizes, int n_in,
                              void* d_out, int out_size, void* d_ws,
                              size_t ws_size, hipStream_t stream) {
  const float* q    = (const float*)d_in[0];
  const float* k    = (const float*)d_in[1];
  const float* v    = (const float*)d_in[2];
  const float* wq_w = (const float*)d_in[3];
  const float* wq_b = (const float*)d_in[4];
  const float* wk_w = (const float*)d_in[5];
  const float* wk_b = (const float*)d_in[6];
  const float* wv_w = (const float*)d_in[7];
  const float* wv_b = (const float*)d_in[8];
  const float* wo_w = (const float*)d_in[9];
  const float* wo_b = (const float*)d_in[10];

  float* out  = (float*)d_out;
  float* attn = out + (size_t)MROWS * D_MODEL;  // tuple: (out, attn)

  char* ws = (char*)d_ws;
  __bf16* Qh     = (__bf16*)(ws);                       // 4 MB
  __bf16* Kh     = (__bf16*)(ws + (4u << 20));          // 4 MB
  __bf16* VhT    = (__bf16*)(ws + (8u << 20));          // 4 MB
  float*  concat = (float*)(ws + (12u << 20));          // 8 MB

  qkv_proj_kernel<<<dim3(768), dim3(256), 0, stream>>>(
      q, k, v, wq_w, wq_b, wk_w, wk_b, wv_w, wv_b, Qh, Kh, VhT);
  attn_kernel<<<dim3(BATCH * NHEAD * (SEQ / 16)), dim3(128), 0, stream>>>(
      Qh, Kh, VhT, attn, concat);
  out_proj_kernel<<<dim3(256), dim3(256), 0, stream>>>(
      concat, wo_w, wo_b, out);
}
